// GroupedExperts_73220602462485
// MI455X (gfx1250) — matrix operand order repaired and checked by reading, not compiled, not run on hardware
//
#include <hip/hip_runtime.h>
#include <hip/hip_bf16.h>

// Grouped-expert SwiGLU MLP for MI455X (gfx1250), wave32 + WMMA bf16.
//   B=4, T=4096, DIM=2048, HID=1024, E=8
// Kernel 1: h = silu(x@w1[e]^T) * (x@w3[e]^T)  -> bf16 in d_ws (needs 32 MiB)
// Kernel 2: y = h@w2[e]^T                      -> f32 in d_out
// Tokens are contiguous per expert along T; each block resolves its expert
// range from the cumsum of num_tokens_per_expert and masks rows at expert
// boundaries (exact searchsorted(side="right") semantics).

typedef __attribute__((ext_vector_type(16))) __bf16 v16bf;
typedef __attribute__((ext_vector_type(8)))  __bf16 v8bf;
typedef __attribute__((ext_vector_type(8)))  float  v8f;

#define B_   4
#define T_   4096
#define DIM_ 2048
#define HID_ 1024
#define E_   8

#define BM  128   // token tile
#define BN  128   // output-column tile
#define BK  32    // K step (matches 16x16x32 bf16 WMMA)
#define LDA 40    // padded LDS row stride (bf16 elems) to spread banks

__device__ __forceinline__ int expert_of(const int* cum, int t) {
  int e = 0;
#pragma unroll
  for (int i = 0; i < E_; ++i) e += (t >= cum[i]) ? 1 : 0;
  return e;  // first e with cum[e] > t
}

__device__ __forceinline__ v16bf join8(v8bf lo, v8bf hi) {
  return __builtin_shufflevector(lo, hi, 0, 1, 2, 3, 4, 5, 6, 7,
                                 8, 9, 10, 11, 12, 13, 14, 15);
}

// Stage a 128-row x 32-col f32 slab into LDS as bf16. 256 threads: 2/row.
__device__ __forceinline__ void load_tile_f32(__bf16* sdst, const float* src,
                                              int srcStride, int k0, int tid) {
  const int row = tid >> 1;
  const int kh  = (tid & 1) << 4;  // 0 or 16
  const float4* p = (const float4*)(src + (size_t)row * srcStride + k0 + kh);
  union { __bf16 h[16]; uint4 q[2]; } u;
#pragma unroll
  for (int j = 0; j < 4; ++j) {
    float4 f = p[j];
    u.h[4 * j + 0] = (__bf16)f.x;
    u.h[4 * j + 1] = (__bf16)f.y;
    u.h[4 * j + 2] = (__bf16)f.z;
    u.h[4 * j + 3] = (__bf16)f.w;
  }
  uint4* d = (uint4*)(sdst + (size_t)row * LDA + kh);
  d[0] = u.q[0];
  d[1] = u.q[1];
}

// Stage a 128-row x 32-col bf16 slab into LDS (no conversion).
__device__ __forceinline__ void load_tile_bf16(__bf16* sdst, const __bf16* src,
                                               int srcStride, int k0, int tid) {
  const int row = tid >> 1;
  const int kh  = (tid & 1) << 4;
  const uint4* p = (const uint4*)(src + (size_t)row * srcStride + k0 + kh);
  uint4* d = (uint4*)(sdst + (size_t)row * LDA + kh);
  d[0] = p[0];
  d[1] = p[1];
}

// ---------------- Kernel 1: fused gate/up + SwiGLU ----------------
__global__ __launch_bounds__(256) void moe_gate_up_kernel(
    const float* __restrict__ x, const float* __restrict__ w1,
    const float* __restrict__ w3, const int* __restrict__ counts,
    __bf16* __restrict__ hbuf) {
  __shared__ __bf16 sA[BM * LDA];
  __shared__ __bf16 sB1[BN * LDA];
  __shared__ __bf16 sB3[BN * LDA];

  const int tid  = threadIdx.x;
  const int lane = tid & 31;
  const int wave = tid >> 5;
  const int fr   = lane & 15;  // fragment row/col within 16
  const int fh   = lane >> 4;  // which half of the wave

  int bid = blockIdx.x;
  const int nb = bid % (HID_ / BN); bid /= (HID_ / BN);
  const int tb = bid % (T_ / BM);
  const int b  = bid / (T_ / BM);
  const int t0 = tb * BM;
  const int n0 = nb * BN;

  int cum[E_];
  {
    const int* c = counts + b * E_;
    int s = 0;
#pragma unroll
    for (int i = 0; i < E_; ++i) { s += c[i]; cum[i] = s; }
  }
  const int eLo = expert_of(cum, t0);
  const int eHi = expert_of(cum, t0 + BM - 1);

  const int mbase = (wave & 3) * 32;   // 4 wave-rows of 32 tokens
  const int nbase = (wave >> 2) * 64;  // 2 wave-cols of 64 columns

  const float* xblk = x + ((size_t)b * T_ + t0) * DIM_;

  for (int e = eLo; e <= eHi; ++e) {
    const float* w1e = w1 + (size_t)e * HID_ * DIM_ + (size_t)n0 * DIM_;
    const float* w3e = w3 + (size_t)e * HID_ * DIM_ + (size_t)n0 * DIM_;

    v8f accg[2][4], accu[2][4];
#pragma unroll
    for (int mt = 0; mt < 2; ++mt)
#pragma unroll
      for (int nt = 0; nt < 4; ++nt) {
        accg[mt][nt] = (v8f)(0.0f);
        accu[mt][nt] = (v8f)(0.0f);
      }

    for (int k0 = 0; k0 < DIM_; k0 += BK) {
      load_tile_f32(sA, xblk, DIM_, k0, tid);
      load_tile_f32(sB1, w1e, DIM_, k0, tid);
      load_tile_f32(sB3, w3e, DIM_, k0, tid);
      if (k0 + BK < DIM_) {  // emits global_prefetch_b8
        __builtin_prefetch(xblk + (size_t)(tid >> 1) * DIM_ + k0 + BK, 0, 0);
        __builtin_prefetch(w1e + (size_t)(tid >> 1) * DIM_ + k0 + BK, 0, 0);
        __builtin_prefetch(w3e + (size_t)(tid >> 1) * DIM_ + k0 + BK, 0, 0);
      }
      __syncthreads();

      v16bf afr[2], b1f[4], b3f[4];
#pragma unroll
      for (int mt = 0; mt < 2; ++mt) {
        const __bf16* ap = sA + (size_t)(mbase + 16 * mt + fr) * LDA + 16 * fh;
        afr[mt] = join8(*(const v8bf*)ap, *(const v8bf*)(ap + 8));
      }
#pragma unroll
      for (int nt = 0; nt < 4; ++nt) {
        const __bf16* bp1 = sB1 + (size_t)(nbase + 16 * nt + fr) * LDA + 16 * fh;
        b1f[nt] = join8(*(const v8bf*)bp1, *(const v8bf*)(bp1 + 8));
        const __bf16* bp3 = sB3 + (size_t)(nbase + 16 * nt + fr) * LDA + 16 * fh;
        b3f[nt] = join8(*(const v8bf*)bp3, *(const v8bf*)(bp3 + 8));
      }
#pragma unroll
      for (int mt = 0; mt < 2; ++mt)
#pragma unroll
        for (int nt = 0; nt < 4; ++nt) {
          accg[mt][nt] = __builtin_amdgcn_wmma_f32_16x16x32_bf16(
              false, afr[mt], false, b1f[nt], (short)0, accg[mt][nt], false, false);
          accu[mt][nt] = __builtin_amdgcn_wmma_f32_16x16x32_bf16(
              false, afr[mt], false, b3f[nt], (short)0, accu[mt][nt], false, false);
        }
      __syncthreads();
    }

    // SwiGLU epilogue + masked bf16 store (only rows owned by expert e).
    const int lo = (e == 0) ? 0 : cum[e - 1];
    const int hi = cum[e];
#pragma unroll
    for (int mt = 0; mt < 2; ++mt) {
#pragma unroll
      for (int r = 0; r < 8; ++r) {
        const int tok = t0 + mbase + 16 * mt + 8 * fh + r;
        const bool act = (tok >= lo) && (tok < hi);
#pragma unroll
        for (int nt = 0; nt < 4; ++nt) {
          const float g = accg[mt][nt][r];
          const float u = accu[mt][nt][r];
          const float hval = (g / (1.0f + __expf(-g))) * u;  // silu(g)*u
          if (act) {
            hbuf[((size_t)b * T_ + tok) * HID_ + n0 + nbase + 16 * nt + fr] =
                (__bf16)hval;
          }
        }
      }
    }
  }
}

// ---------------- Kernel 2: down projection ----------------
__global__ __launch_bounds__(256) void moe_down_kernel(
    const __bf16* __restrict__ hbuf, const float* __restrict__ w2,
    const int* __restrict__ counts, float* __restrict__ out) {
  __shared__ __bf16 sA[BM * LDA];
  __shared__ __bf16 sB[BN * LDA];

  const int tid  = threadIdx.x;
  const int lane = tid & 31;
  const int wave = tid >> 5;
  const int fr   = lane & 15;
  const int fh   = lane >> 4;

  int bid = blockIdx.x;
  const int nb = bid % (DIM_ / BN); bid /= (DIM_ / BN);
  const int tb = bid % (T_ / BM);
  const int b  = bid / (T_ / BM);
  const int t0 = tb * BM;
  const int n0 = nb * BN;

  int cum[E_];
  {
    const int* c = counts + b * E_;
    int s = 0;
#pragma unroll
    for (int i = 0; i < E_; ++i) { s += c[i]; cum[i] = s; }
  }
  const int eLo = expert_of(cum, t0);
  const int eHi = expert_of(cum, t0 + BM - 1);

  const int mbase = (wave & 3) * 32;
  const int nbase = (wave >> 2) * 64;

  const __bf16* hblk = hbuf + ((size_t)b * T_ + t0) * HID_;

  for (int e = eLo; e <= eHi; ++e) {
    const float* w2e = w2 + (size_t)e * DIM_ * HID_ + (size_t)n0 * HID_;

    v8f acc[2][4];
#pragma unroll
    for (int mt = 0; mt < 2; ++mt)
#pragma unroll
      for (int nt = 0; nt < 4; ++nt) acc[mt][nt] = (v8f)(0.0f);

    for (int k0 = 0; k0 < HID_; k0 += BK) {
      load_tile_bf16(sA, hblk, HID_, k0, tid);
      load_tile_f32(sB, w2e, HID_, k0, tid);
      if (k0 + BK < HID_) {
        __builtin_prefetch(hblk + (size_t)(tid >> 1) * HID_ + k0 + BK, 0, 0);
        __builtin_prefetch(w2e + (size_t)(tid >> 1) * HID_ + k0 + BK, 0, 0);
      }
      __syncthreads();

      v16bf afr[2], bfr[4];
#pragma unroll
      for (int mt = 0; mt < 2; ++mt) {
        const __bf16* ap = sA + (size_t)(mbase + 16 * mt + fr) * LDA + 16 * fh;
        afr[mt] = join8(*(const v8bf*)ap, *(const v8bf*)(ap + 8));
      }
#pragma unroll
      for (int nt = 0; nt < 4; ++nt) {
        const __bf16* bp = sB + (size_t)(nbase + 16 * nt + fr) * LDA + 16 * fh;
        bfr[nt] = join8(*(const v8bf*)bp, *(const v8bf*)(bp + 8));
      }
#pragma unroll
      for (int mt = 0; mt < 2; ++mt)
#pragma unroll
        for (int nt = 0; nt < 4; ++nt)
          acc[mt][nt] = __builtin_amdgcn_wmma_f32_16x16x32_bf16(
              false, afr[mt], false, bfr[nt], (short)0, acc[mt][nt], false, false);
      __syncthreads();
    }

    const int lo = (e == 0) ? 0 : cum[e - 1];
    const int hi = cum[e];
#pragma unroll
    for (int mt = 0; mt < 2; ++mt) {
#pragma unroll
      for (int r = 0; r < 8; ++r) {
        const int tok = t0 + mbase + 16 * mt + 8 * fh + r;
        const bool act = (tok >= lo) && (tok < hi);
#pragma unroll
        for (int nt = 0; nt < 4; ++nt) {
          if (act) {
            out[((size_t)b * T_ + tok) * DIM_ + n0 + nbase + 16 * nt + fr] =
                acc[mt][nt][r];
          }
        }
      }
    }
  }
}

extern "C" void kernel_launch(void* const* d_in, const int* in_sizes, int n_in,
                              void* d_out, int out_size, void* d_ws, size_t ws_size,
                              hipStream_t stream) {
  const float* x      = (const float*)d_in[0];
  const float* w1     = (const float*)d_in[1];
  const float* w2     = (const float*)d_in[2];
  const float* w3     = (const float*)d_in[3];
  const int*   counts = (const int*)d_in[4];
  float* out = (float*)d_out;

  // Intermediate h (bf16): B*T*HID*2 = 32 MiB in workspace.
  __bf16* hbuf = (__bf16*)d_ws;

  dim3 blk(256);
  moe_gate_up_kernel<<<dim3(B_ * (T_ / BM) * (HID_ / BN)), blk, 0, stream>>>(
      x, w1, w3, counts, hbuf);
  moe_down_kernel<<<dim3(B_ * (T_ / BM) * (DIM_ / BN)), blk, 0, stream>>>(
      hbuf, w2, counts, out);
}
